// DualVQGNN_16724602651150
// MI455X (gfx1250) — matrix operand order repaired and checked by reading, not compiled, not run on hardware
//
#include <hip/hip_runtime.h>
#include <hip/hip_bf16.h>

typedef __attribute__((ext_vector_type(16))) _Float16 v16h;
typedef __attribute__((ext_vector_type(8)))  float    v8f;

#define WMMA(a, b, c) __builtin_amdgcn_wmma_f32_16x16x32_f16( \
    false, (a), false, (b), (short)0, (c), false, false)

constexpr int NN     = 50000;
constexpr int NE     = 800000;
constexpr int IN_DIM = 256;
constexpr int HID    = 128;
constexpr int OUTD   = 64;
constexpr int CB1    = 2048;
constexpr int CB2    = 4096;
constexpr int VQD    = 64;
constexpr float COMMIT_W = 0.25f;

// ---------------------------------------------------------------------------
// CDNA5 wave32 WMMA fragment layouts (16x16x32 f16):
// A 16x32: lane m=L%16, kb=(L<16?0:8); a[i<8]=A[m][k0+kb+i], a[8+i]=A[m][k0+kb+16+i]
// B 32x16: lane n=L%16, same k order as A: b[i<8]=B[k0+kb+i][n], b[8+i]=B[k0+kb+16+i][n]
// C/D:     vgpr r <-> row r+8*(L>=16), col = L%16
// Weights are pre-packed so a B fragment is one contiguous 32B per-lane read:
//   Bp[((tn*Kt + kc)*32 + lane)*16 + i]
// ---------------------------------------------------------------------------
__device__ __forceinline__ v16h load_a_h(const _Float16* A, int lda, int m, int k0) {
  const int lane = threadIdx.x & 31;
  const int kb = (lane < 16) ? 0 : 8;
  const _Float16* r = A + (size_t)m * lda + k0 + kb;
  v16h a;
#pragma unroll
  for (int i = 0; i < 8; ++i) {
    a[i]     = r[i];
    a[8 + i] = r[16 + i];
  }
  return a;
}

__device__ __forceinline__ v16h load_b_packed(const _Float16* Bp, int Kt, int tn, int kc) {
  const int lane = threadIdx.x & 31;
  return *(const v16h*)(Bp + (((size_t)tn * Kt + kc) * 32 + lane) * 16);
}

// ---------------------------------------------------------------------------
// Utility kernels
// ---------------------------------------------------------------------------
__global__ void k_zero(float* p, int n) {
  int i = blockIdx.x * blockDim.x + threadIdx.x;
  if (i < n) p[i] = 0.0f;
}

__global__ void k_f32_to_f16(const float* __restrict__ in, _Float16* __restrict__ out,
                             int n) {
  int i = blockIdx.x * blockDim.x + threadIdx.x;
  if (i < n) out[i] = (_Float16)in[i];
}

// pack B[K,N] (f32 row-major) into f16 fragment-major layout
__global__ void k_pack_b(const float* __restrict__ B, _Float16* __restrict__ Bp,
                         int K, int N) {
  int t = blockIdx.x * blockDim.x + threadIdx.x;
  if (t >= K * N) return;
  int i    = t & 15;
  int lane = (t >> 4) & 31;
  int fc   = t >> 9;                   // tn*Kt + kc
  int Kt   = K >> 5;
  int kc   = fc % Kt;
  int tn   = fc / Kt;
  int n    = lane & 15;
  int kb   = (lane < 16) ? 0 : 8;
  Bp[t] = (_Float16)B[(size_t)(kc * 32 + kb + (i < 8 ? i : 8 + i)) * N + tn * 16 + n];
}

__global__ void k_degrees(const int* __restrict__ src, const int* __restrict__ dst,
                          float* dout, float* din, int ne) {
  int e = blockIdx.x * blockDim.x + threadIdx.x;
  if (e < ne) {
    atomicAdd(&dout[src[e]], 1.0f);
    atomicAdd(&din[dst[e]], 1.0f);
  }
}

__global__ void k_deg_inv(float* d, int n) {
  int i = blockIdx.x * blockDim.x + threadIdx.x;
  if (i < n) d[i] = rsqrtf(fmaxf(d[i], 1.0f));
}

// agg[dst[e]][c] += h[src[e]][c]   (dim = 1<<shift)
__global__ void k_edge_agg(const float* __restrict__ h, const int* __restrict__ src,
                           const int* __restrict__ dst, float* agg, int ne, int shift) {
  long long i = (long long)blockIdx.x * blockDim.x + threadIdx.x;
  int e = (int)(i >> shift);
  if (e >= ne) return;
  int c = (int)(i & ((1 << shift) - 1));
  atomicAdd(&agg[((size_t)dst[e] << shift) + c], h[((size_t)src[e] << shift) + c]);
}

// x = x*din_inv[row] + bias[col]  (in place, f32) — used for the final conv output
__global__ void k_postagg_f32(float* x, const float* __restrict__ din_inv,
                              const float* __restrict__ bias, int total, int shift) {
  int i = blockIdx.x * blockDim.x + threadIdx.x;
  if (i >= total) return;
  int row = i >> shift;
  int c = i & ((1 << shift) - 1);
  x[i] = x[i] * din_inv[row] + bias[c];
}

// out_f16 = relu(agg*din_inv[row] + bias[col]) — conv1 epilogue, f16 for next GEMM
__global__ void k_postagg_relu_f16(const float* __restrict__ agg,
                                   _Float16* __restrict__ out,
                                   const float* __restrict__ din_inv,
                                   const float* __restrict__ bias, int total, int shift) {
  int i = blockIdx.x * blockDim.x + threadIdx.x;
  if (i >= total) return;
  int row = i >> shift;
  int c = i & ((1 << shift) - 1);
  out[i] = (_Float16)fmaxf(agg[i] * din_inv[row] + bias[c], 0.0f);
}

// L2-normalize codebook rows (VQD=64), store f16. One wave per row.
__global__ void k_norm_codebook(const float* __restrict__ E, _Float16* Eh, int rows) {
  int wave = (blockIdx.x * blockDim.x + threadIdx.x) >> 5;
  if (wave >= rows) return;
  int lane = threadIdx.x & 31;
  const float* r = E + (size_t)wave * VQD;
  float a = r[lane], b = r[lane + 32];
  float s = a * a + b * b;
#pragma unroll
  for (int off = 16; off; off >>= 1) s += __shfl_xor(s, off, 32);
  float rn = rsqrtf(s + 1e-12f);
  _Float16* o = Eh + (size_t)wave * VQD;
  o[lane]      = (_Float16)(a * rn);
  o[lane + 32] = (_Float16)(b * rn);
}

// ---------------------------------------------------------------------------
// WMMA GEMM: each wave computes a 16x64 strip of C = (A@B)*rowscale[m].
// A: f16 row-major [M,K]; Bp: packed f16 fragments; M%16==0, N%64==0, K%32==0.
// ---------------------------------------------------------------------------
__global__ void k_wmma_gemm(const _Float16* __restrict__ A, const _Float16* __restrict__ Bp,
                            float* __restrict__ C, int M, int N, int K,
                            const float* __restrict__ rowscale) {
  int wave = (blockIdx.x * blockDim.x + threadIdx.x) >> 5;
  int ng = N >> 6;                       // 64-col strips
  int tm = wave / ng, tg = wave % ng;
  if (tm * 16 >= M) return;
  int lane = threadIdx.x & 31;
  int m = tm * 16 + (lane & 15);
  int Kt = K >> 5;
  v8f acc[4] = {};
  for (int kc = 0; kc < Kt; ++kc) {
    v16h a = load_a_h(A, K, m, kc * 32);
#pragma unroll
    for (int t = 0; t < 4; ++t) {
      v16h b = load_b_packed(Bp, Kt, tg * 4 + t, kc);
      acc[t] = WMMA(a, b, acc[t]);
    }
  }
  int rbase = tm * 16 + ((lane >> 4) << 3);
#pragma unroll
  for (int t = 0; t < 4; ++t) {
    int col = tg * 64 + t * 16 + (lane & 15);
#pragma unroll
    for (int r = 0; r < 8; ++r) {
      int row = rbase + r;
      C[(size_t)row * N + col] = acc[t][r] * rowscale[row];
    }
  }
}

// ---------------------------------------------------------------------------
// VQ: fused l2norm(h) + xn @ en^T (WMMA) + running argmax.
// One wave handles 64 rows (4 A pairs); each codebook tile feeds 8 WMMAs.
// ---------------------------------------------------------------------------
__global__ void k_vq_argmax(const float* __restrict__ H, const _Float16* __restrict__ Eh,
                            int cb, int* __restrict__ idx) {
  int wave = (blockIdx.x * blockDim.x + threadIdx.x) >> 5;
  int base = wave * 64;
  if (base >= NN) return;
  int lane = threadIdx.x & 31;
  int kb = (lane < 16) ? 0 : 16;

  v16h a0[4], a1[4];
#pragma unroll
  for (int g = 0; g < 4; ++g) {
    int m = base + g * 16 + (lane & 15);
    if (m >= NN) m = NN - 1;             // tail: duplicate last row (writes guarded)
    const float* row = H + (size_t)m * VQD;
    float r0[16], r1[16];
#pragma unroll
    for (int i = 0; i < 8; ++i) {
      r0[i] = row[kb + i];       r0[8 + i] = row[kb + 8 + i];
      r1[i] = row[32 + kb + i];  r1[8 + i] = row[32 + kb + 8 + i];
    }
    float s = 0.0f;
#pragma unroll
    for (int i = 0; i < 16; ++i) s += r0[i] * r0[i] + r1[i] * r1[i];
    s += __shfl_xor(s, 16, 32);          // lanes L and L^16 hold the two row halves
    float rn = rsqrtf(s + 1e-12f);
#pragma unroll
    for (int i = 0; i < 16; ++i) {
      a0[g][i] = (_Float16)(r0[i] * rn);
      a1[g][i] = (_Float16)(r1[i] * rn);
    }
  }

  float bestv[4][8];
  int   besti[4][8];
#pragma unroll
  for (int g = 0; g < 4; ++g)
#pragma unroll
    for (int r = 0; r < 8; ++r) { bestv[g][r] = -3.4e38f; besti[g][r] = 0; }

  const int koff = (lane < 16) ? 0 : 16;
  for (int jt = 0; jt < (cb >> 4); ++jt) {
    const _Float16* er = Eh + (size_t)(jt * 16 + (lane & 15)) * VQD + koff;
    v16h b0 = *(const v16h*)(er);
    v16h b1 = *(const v16h*)(er + 32);
    int cidx = jt * 16 + (lane & 15);
#pragma unroll
    for (int g = 0; g < 4; ++g) {
      v8f acc = {};
      acc = WMMA(a0[g], b0, acc);
      acc = WMMA(a1[g], b1, acc);
#pragma unroll
      for (int r = 0; r < 8; ++r) {
        if (acc[r] > bestv[g][r]) { bestv[g][r] = acc[r]; besti[g][r] = cidx; }
      }
    }
  }
  // argmax reduce across the 16-lane column group (stays within half-wave)
#pragma unroll
  for (int off = 1; off < 16; off <<= 1) {
#pragma unroll
    for (int g = 0; g < 4; ++g) {
#pragma unroll
      for (int r = 0; r < 8; ++r) {
        float ov = __shfl_xor(bestv[g][r], off, 32);
        int   oi = __shfl_xor(besti[g][r], off, 32);
        if (ov > bestv[g][r] || (ov == bestv[g][r] && oi < besti[g][r])) {
          bestv[g][r] = ov; besti[g][r] = oi;
        }
      }
    }
  }
  if ((lane & 15) == 0) {
    int rb = (lane >> 4) << 3;
#pragma unroll
    for (int g = 0; g < 4; ++g) {
#pragma unroll
      for (int r = 0; r < 8; ++r) {
        int row = base + g * 16 + rb + r;
        if (row < NN) idx[row] = besti[g][r];
      }
    }
  }
}

// q(f16) = en[idx]; accumulate COMMIT_W * mean((q - h)^2)
__global__ void k_gather_commit(const float* __restrict__ H, const _Float16* __restrict__ Eh,
                                const int* __restrict__ idx, _Float16* __restrict__ qh,
                                float* loss_out, float scale) {
  __shared__ float sred[8];
  int i = blockIdx.x * blockDim.x + threadIdx.x;
  float part = 0.0f;
  if (i < NN * VQD) {
    int n = i >> 6, d = i & 63;
    _Float16 qv = Eh[(size_t)idx[n] * VQD + d];
    qh[i] = qv;
    float diff = (float)qv - H[i];
    part = diff * diff;
  }
#pragma unroll
  for (int off = 16; off; off >>= 1) part += __shfl_xor(part, off, 32);
  int lane = threadIdx.x & 31, w = threadIdx.x >> 5;
  if (lane == 0) sred[w] = part;
  __syncthreads();
  if (threadIdx.x == 0) {
    float ssum = 0.0f;
#pragma unroll
    for (int k = 0; k < 8; ++k) ssum += sred[k];
    atomicAdd(loss_out, ssum * scale);
  }
}

// ---------------------------------------------------------------------------
// Fused head MLP: P = relu(Q @ w1 + b1) @ w2 + b2
// Q: f16 [NN,64]; w1p/w2p: packed fragments. One wave (WG of 32) per 16 rows;
// hidden re-laid out C-format -> A-format via LDS (f16).
// ---------------------------------------------------------------------------
__global__ void k_head_mlp(const _Float16* __restrict__ Q,
                           const _Float16* __restrict__ w1p, const float* __restrict__ b1,
                           const _Float16* __restrict__ w2p, const float* __restrict__ b2,
                           float* __restrict__ P) {
  __shared__ _Float16 hl[16 * 128];
  int wave = blockIdx.x;                 // blockDim.x == 32
  if (wave * 16 >= NN) return;
  int lane = threadIdx.x & 31;
  int m = wave * 16 + (lane & 15);
  int nn = lane & 15;
  int rb = (lane >> 4) << 3;

  v16h aq0 = load_a_h(Q, VQD, m, 0);
  v16h aq1 = load_a_h(Q, VQD, m, 32);

  // layer 1: hidden[16][128] = relu(Q@w1 + b1) -> LDS as f16   (w1: Kt=2, Nt=8)
#pragma unroll
  for (int j = 0; j < 8; ++j) {
    float bb = b1[j * 16 + nn];
    v8f acc;
#pragma unroll
    for (int r = 0; r < 8; ++r) acc[r] = bb;
    acc = WMMA(aq0, load_b_packed(w1p, 2, j, 0), acc);
    acc = WMMA(aq1, load_b_packed(w1p, 2, j, 1), acc);
#pragma unroll
    for (int r = 0; r < 8; ++r)
      hl[(rb + r) * 128 + j * 16 + nn] = (_Float16)fmaxf(acc[r], 0.0f);
  }
  __syncthreads();

  // layer 2: P = hidden @ w2 + b2   (w2: Kt=4, Nt=4)
  const int akb = (lane < 16) ? 0 : 8;
#pragma unroll
  for (int jo = 0; jo < 4; ++jo) {
    float bb = b2[jo * 16 + nn];
    v8f acc;
#pragma unroll
    for (int r = 0; r < 8; ++r) acc[r] = bb;
#pragma unroll
    for (int kc = 0; kc < 4; ++kc) {
      const _Float16* hr = hl + (lane & 15) * 128 + kc * 32 + akb;
      v16h a;
#pragma unroll
      for (int i = 0; i < 8; ++i) { a[i] = hr[i]; a[8 + i] = hr[16 + i]; }
      acc = WMMA(a, load_b_packed(w2p, 4, jo, kc), acc);
    }
    int rowb = wave * 16 + rb;
#pragma unroll
    for (int r = 0; r < 8; ++r)
      P[(size_t)(rowb + r) * OUTD + jo * 16 + nn] = acc[r];
  }
}

// ---------------------------------------------------------------------------
// Host launcher
// ---------------------------------------------------------------------------
static inline int gb(long long total, int bs) { return (int)((total + bs - 1) / bs); }

extern "C" void kernel_launch(void* const* d_in, const int* in_sizes, int n_in,
                              void* d_out, int out_size, void* d_ws, size_t ws_size,
                              hipStream_t stream) {
  const float* feats  = (const float*)d_in[0];
  const int*   src    = (const int*)d_in[1];
  const int*   dst    = (const int*)d_in[2];
  const float* W1     = (const float*)d_in[3];
  const float* b1     = (const float*)d_in[4];
  const float* W2     = (const float*)d_in[5];
  const float* b2     = (const float*)d_in[6];
  const float* embed1 = (const float*)d_in[7];
  const float* embed2 = (const float*)d_in[8];
  const float* p1w1 = (const float*)d_in[9],  *p1b1 = (const float*)d_in[10];
  const float* p1w2 = (const float*)d_in[11], *p1b2 = (const float*)d_in[12];
  const float* p2w1 = (const float*)d_in[13], *p2b1 = (const float*)d_in[14];
  const float* p2w2 = (const float*)d_in[15], *p2b2 = (const float*)d_in[16];

  float* out = (float*)d_out;
  float* p1  = out;
  float* p2  = out + (size_t)NN * OUTD;
  float* losses = out + (size_t)2 * NN * OUTD;   // [loss1, loss2]

  // workspace layout (all regions 32B aligned)
  float* W = (float*)d_ws;
  float* deg_out_inv = W;                              // NN
  float* deg_in_inv  = W + NN;                         // NN
  float* bufA = W + 2 * NN;                            // NN*HID f32: h0 -> {h2pre|agg2/h}
  float* bufB = bufA + (size_t)NN * HID;               // NN*HID f32: agg1
  _Float16* featsh = (_Float16*)(bufB + (size_t)NN * HID);  // NN*IN_DIM f16 (later qh1,qh2)
  _Float16* h1h  = featsh + (size_t)NN * IN_DIM;       // NN*HID f16
  _Float16* enh  = h1h + (size_t)NN * HID;             // (CB1+CB2)*VQD f16
  _Float16* enh1 = enh;
  _Float16* enh2 = enh + (size_t)CB1 * VQD;
  _Float16* wpack = enh + (size_t)(CB1 + CB2) * VQD;   // 73728 f16 packed weights
  _Float16* W1p   = wpack;                             // 32768
  _Float16* W2p   = W1p + 32768;                       // 8192
  _Float16* p1w1p = W2p + 8192;                        // 8192
  _Float16* p1w2p = p1w1p + 8192;                      // 8192
  _Float16* p2w1p = p1w2p + 8192;                      // 8192
  _Float16* p2w2p = p2w1p + 8192;                      // 8192
  int* idx1 = (int*)(wpack + 73728);
  int* idx2 = idx1 + NN;

  float* h2pre = bufA;                                 // NN*OUTD f32
  float* hfin  = bufA + (size_t)NN * OUTD;             // NN*OUTD f32 (agg2 -> h)
  _Float16* qh1 = featsh;                              // NN*VQD f16 (featsh dead by then)
  _Float16* qh2 = featsh + (size_t)NN * VQD;

  const int BS = 256;

  // zero accumulators + loss slots
  k_zero<<<gb(2 * NN, BS), BS, 0, stream>>>(deg_out_inv, 2 * NN);
  k_zero<<<gb((long long)NN * HID, BS), BS, 0, stream>>>(bufB, NN * HID);
  k_zero<<<1, 32, 0, stream>>>(losses, 2);

  // degrees
  k_degrees<<<gb(NE, BS), BS, 0, stream>>>(src, dst, deg_out_inv, deg_in_inv, NE);
  k_deg_inv<<<gb(2 * NN, BS), BS, 0, stream>>>(deg_out_inv, 2 * NN);

  // pre-pack weights into fragment-major f16; convert feats to f16
  k_pack_b<<<gb(IN_DIM * HID, BS), BS, 0, stream>>>(W1, W1p, IN_DIM, HID);
  k_pack_b<<<gb(HID * OUTD, BS), BS, 0, stream>>>(W2, W2p, HID, OUTD);
  k_pack_b<<<gb(VQD * HID, BS), BS, 0, stream>>>(p1w1, p1w1p, VQD, HID);
  k_pack_b<<<gb(HID * OUTD, BS), BS, 0, stream>>>(p1w2, p1w2p, HID, OUTD);
  k_pack_b<<<gb(VQD * HID, BS), BS, 0, stream>>>(p2w1, p2w1p, VQD, HID);
  k_pack_b<<<gb(HID * OUTD, BS), BS, 0, stream>>>(p2w2, p2w2p, HID, OUTD);
  k_f32_to_f16<<<gb((long long)NN * IN_DIM, BS), BS, 0, stream>>>(feats, featsh,
                                                                  NN * IN_DIM);

  // conv1: h0 = (feats@W1)*deg_out^-.5 ; edge agg ; relu(*deg_in^-.5 + b1) -> f16
  {
    long long waves = (long long)(NN / 16) * (HID / 64);
    k_wmma_gemm<<<gb(waves * 32, BS), BS, 0, stream>>>(featsh, W1p, bufA, NN, HID, IN_DIM,
                                                       deg_out_inv);
  }
  k_edge_agg<<<gb((long long)NE << 7, BS), BS, 0, stream>>>(bufA, src, dst, bufB, NE, 7);
  k_postagg_relu_f16<<<gb((long long)NN * HID, BS), BS, 0, stream>>>(
      bufB, h1h, deg_in_inv, b1, NN * HID, 7);

  // conv2
  k_zero<<<gb((long long)NN * OUTD, BS), BS, 0, stream>>>(hfin, NN * OUTD);
  {
    long long waves = (long long)(NN / 16) * (OUTD / 64);
    k_wmma_gemm<<<gb(waves * 32, BS), BS, 0, stream>>>(h1h, W2p, h2pre, NN, OUTD, HID,
                                                       deg_out_inv);
  }
  k_edge_agg<<<gb((long long)NE << 6, BS), BS, 0, stream>>>(h2pre, src, dst, hfin, NE, 6);
  k_postagg_f32<<<gb((long long)NN * OUTD, BS), BS, 0, stream>>>(hfin, deg_in_inv, b2,
                                                                 NN * OUTD, 6);

  // VQ codebooks (normalize once to f16)
  k_norm_codebook<<<gb((long long)CB1 * 32, BS), BS, 0, stream>>>(embed1, enh1, CB1);
  k_norm_codebook<<<gb((long long)CB2 * 32, BS), BS, 0, stream>>>(embed2, enh2, CB2);

  // argmax over cosine similarity (WMMA, 64 rows/wave)
  {
    long long waves = (NN + 63) / 64;
    k_vq_argmax<<<gb(waves * 32, BS), BS, 0, stream>>>(hfin, enh1, CB1, idx1);
    k_vq_argmax<<<gb(waves * 32, BS), BS, 0, stream>>>(hfin, enh2, CB2, idx2);
  }

  // gather q (f16) + commit losses
  const float lscale = COMMIT_W / (float)(NN * VQD);
  k_gather_commit<<<gb((long long)NN * VQD, BS), BS, 0, stream>>>(hfin, enh1, idx1, qh1,
                                                                  &losses[0], lscale);
  k_gather_commit<<<gb((long long)NN * VQD, BS), BS, 0, stream>>>(hfin, enh2, idx2, qh2,
                                                                  &losses[1], lscale);

  // prediction heads (fused 2-layer MLP, WMMA, packed weights)
  k_head_mlp<<<NN / 16, 32, 0, stream>>>(qh1, p1w1p, p1b1, p1w2p, p1b2, p1);
  k_head_mlp<<<NN / 16, 32, 0, stream>>>(qh2, p2w1p, p2b1, p2w2p, p2b2, p2);
}
